// our_model_59158879535675
// MI455X (gfx1250) — compile-verified, matrix operand order checked by reading, not run on hardware
//
#include <hip/hip_runtime.h>
#include <hip/hip_bf16.h>
#include <math.h>

typedef unsigned short u16;
typedef unsigned int   u32;
typedef __attribute__((ext_vector_type(16))) __bf16 bf16x16;
typedef __attribute__((ext_vector_type(4)))  u32    u32x4;
typedef __attribute__((ext_vector_type(8)))  float  f32x8;

#define DEV __device__ __forceinline__

DEV u16 f2bf(float f) {                       // fp32 -> bf16 round-to-nearest-even
    u32 u = __builtin_bit_cast(u32, f);
    u += 0x7fffu + ((u >> 16) & 1u);
    return (u16)(u >> 16);
}
DEV float siluf(float x) { return x * (1.0f / (1.0f + __expf(-x))); }
DEV f32x8 zero8() {
    f32x8 v;
    #pragma unroll
    for (int i = 0; i < 8; ++i) v[i] = 0.0f;
    return v;
}
union frag_u { u32x4 q[2]; bf16x16 f; };

// ---------------------------------------------------------------------------
// Generic f32 -> bf16 pack with row/col zero padding.
// out is (Rdst x ldd), in is (Rsrc x lds) reading Csrc valid cols.
// ---------------------------------------------------------------------------
__global__ void k_pack(const float* __restrict__ in, int lds, int Rsrc, int Csrc,
                       u16* __restrict__ out, int ldd, int total) {
    int i = blockIdx.x * 256 + threadIdx.x;
    if (i >= total) return;
    int r = i / ldd, c = i % ldd;
    float v = (r < Rsrc && c < Csrc) ? in[(size_t)r * lds + c] : 0.0f;
    out[i] = f2bf(v);
}
__global__ void k_aneg(const float* __restrict__ in, float* __restrict__ out, int n) {
    int i = blockIdx.x * 256 + threadIdx.x;
    if (i < n) out[i] = -__expf(in[i]);
}

// ---------------------------------------------------------------------------
// Encoder: conv1d(x,(253,45,2),stride2) + bias + SiLU -> bf16 A-matrix
// rows = (b*253 + c), lda = 3360 (cols 3336..3359 zero-padded)
// ---------------------------------------------------------------------------
__global__ void k_enc(const float* __restrict__ x, const float* __restrict__ w,
                      const float* __restrict__ bias, u16* __restrict__ out) {
    const int NF = 3336, NFP = 3360, TS = 253;
    int i = blockIdx.x * 256 + threadIdx.x;
    if (i >= 32 * TS * NFP) return;
    int n = i % NFP; int c = (i / NFP) % TS; int b = i / (NFP * TS);
    if (n >= NF) { out[i] = 0; return; }
    const float* xp = x + (size_t)b * 45 * 6672 + 2 * n;
    const float* wp = w + c * 90;
    float s = bias[c];
    #pragma unroll 5
    for (int j = 0; j < 45; ++j)
        s += xp[j * 6672] * wp[2 * j] + xp[j * 6672 + 1] * wp[2 * j + 1];
    out[i] = f2bf(siluf(s));
}

// ---------------------------------------------------------------------------
// Batched WMMA bf16 GEMM, both operands K-major: C[z][M,N] = A[z][M,K]*B[z][N,K]^T
// Requirements (guaranteed by caller-side padding):
//   Kp % 32 == 0; lda,ldb % 8 == 0; A has grid.y*64 rows; B has grid.x*128 rows.
// Software-pipelined: fragments for step k+1 are loaded before the 4 WMMAs of
// step k, so WMMAs overlap with in-flight b128 loads (partial loadcnt waits).
// EPI: 0 = linear, 1 = sigmoid, 2 = softplus   (biasN per-col, biasM per-row)
// ---------------------------------------------------------------------------
template<int EPI>
__launch_bounds__(256)
__global__ void k_gemm(const u16* __restrict__ A, int lda, size_t strideA,
                       const u16* __restrict__ Bw, int ldb, size_t strideB,
                       float* __restrict__ C, int ldc, size_t strideC,
                       u16* __restrict__ Cb,
                       const float* __restrict__ biasN,
                       const float* __restrict__ biasM,
                       int M, int N, int Kp) {
    const int lane  = threadIdx.x & 31;
    const int wave  = threadIdx.x >> 5;
    const int m0    = blockIdx.y * 64 + (wave >> 1) * 16;
    const int n0    = blockIdx.x * 128 + (wave & 1) * 64;
    const int nlo   = lane & 15;
    const int khalf = (lane >> 4) << 3;   // ISA 16-bit A/B 16x32 lane-half K split

    const u16* ap = A + (size_t)blockIdx.z * strideA + (size_t)(m0 + nlo) * lda + khalf;
    const u16* bp = Bw + (size_t)blockIdx.z * strideB + (size_t)(n0 + nlo) * ldb + khalf;
    const size_t bstep = (size_t)16 * ldb;

    f32x8 acc[4];
    #pragma unroll
    for (int t = 0; t < 4; ++t) acc[t] = zero8();

    // ---- pipeline stage 0 ----
    frag_u a_cur, b_cur[4];
    a_cur.q[0] = *(const u32x4*)(ap);
    a_cur.q[1] = *(const u32x4*)(ap + 16);
    #pragma unroll
    for (int t = 0; t < 4; ++t) {
        const u16* bt = bp + (size_t)t * bstep;
        b_cur[t].q[0] = *(const u32x4*)(bt);
        b_cur[t].q[1] = *(const u32x4*)(bt + 16);
    }

    for (int k0 = 32; k0 < Kp; k0 += 32) {
        ap += 32; bp += 32;
        // issue next-stage loads first ...
        frag_u a_nxt, b_nxt[4];
        a_nxt.q[0] = *(const u32x4*)(ap);
        a_nxt.q[1] = *(const u32x4*)(ap + 16);
        #pragma unroll
        for (int t = 0; t < 4; ++t) {
            const u16* bt = bp + (size_t)t * bstep;
            b_nxt[t].q[0] = *(const u32x4*)(bt);
            b_nxt[t].q[1] = *(const u32x4*)(bt + 16);
        }
        // ... then compute on the current stage while loads are in flight
        #pragma unroll
        for (int t = 0; t < 4; ++t)
            acc[t] = __builtin_amdgcn_wmma_f32_16x16x32_bf16(
                         false, a_cur.f, false, b_cur[t].f, (short)0, acc[t], false, false);
        a_cur = a_nxt;
        #pragma unroll
        for (int t = 0; t < 4; ++t) b_cur[t] = b_nxt[t];
    }
    // ---- drain last stage ----
    #pragma unroll
    for (int t = 0; t < 4; ++t)
        acc[t] = __builtin_amdgcn_wmma_f32_16x16x32_bf16(
                     false, a_cur.f, false, b_cur[t].f, (short)0, acc[t], false, false);

    float* Cp = C + (size_t)blockIdx.z * strideC;
    const int rbase = m0 + khalf;   // C layout: lanes 0-15 rows m0..+7, lanes 16-31 rows +8
    #pragma unroll
    for (int t = 0; t < 4; ++t) {
        int col = n0 + t * 16 + nlo;
        if (col >= N) continue;
        float bn = biasN ? biasN[col] : 0.0f;
        #pragma unroll
        for (int v = 0; v < 8; ++v) {
            int r = rbase + v;
            if (r >= M) continue;
            float xv = acc[t][v] + bn + (biasM ? biasM[r] : 0.0f);
            if (EPI == 1)      xv = 1.0f / (1.0f + __expf(-xv));
            else if (EPI == 2) xv = (xv > 20.0f) ? xv : log1pf(__expf(xv));
            Cp[(size_t)r * ldc + col] = xv;
            if (Cb) Cb[(size_t)r * ldc + col] = f2bf(xv);
        }
    }
}

// ---------------------------------------------------------------------------
// Depthwise conv (kernel 2, stride 2) over W axis, output TRANSPOSED to
// (b, w, ci) with ci padded to 256 (zeros) -> K-major B operand for pw GEMM.
// ---------------------------------------------------------------------------
__global__ void k_dwconvT(const float* __restrict__ x, const float* __restrict__ w2,
                          const float* __restrict__ b, u16* __restrict__ out,
                          int Win, int Wout) {
    int i = blockIdx.x * 256 + threadIdx.x;
    if (i >= 32 * Wout * 256) return;
    int ci = i & 255; int w = (i >> 8) % Wout; int bi = i / (256 * Wout);
    float v = 0.0f;
    if (ci < 253) {
        const float* xp = x + ((size_t)(bi * 253 + ci)) * Win + 2 * w;
        v = xp[0] * w2[ci * 2] + xp[1] * w2[ci * 2 + 1] + b[ci];
    }
    out[i] = f2bf(v);
}

// BatchNorm + SiLU + *2 + positional encoding -> f32 and bf16
__global__ void k_bnpe(const float* __restrict__ in, const float* __restrict__ g,
                       const float* __restrict__ bb, const float* __restrict__ m,
                       const float* __restrict__ v, const float* __restrict__ pe,
                       float* __restrict__ of, u16* __restrict__ ob, int W) {
    int i = blockIdx.x * 256 + threadIdx.x;
    if (i >= 32 * 253 * W) return;
    int wq = i % W; int c = (i / W) % 253;
    float t = (in[i] - m[c]) * (g[c] * rsqrtf(v[c] + 1e-5f)) + bb[c];
    t = 2.0f * siluf(t) + pe[c * W + wq];
    of[i] = t; ob[i] = f2bf(t);
}

// Causal depthwise conv (kernel 4, pad (3,0)) over sequence + SiLU
__global__ void k_cconv(const float* __restrict__ xz, const float* __restrict__ cw,
                        const float* __restrict__ cb, float* __restrict__ of,
                        u16* __restrict__ ob, int di) {
    int i = blockIdx.x * 256 + threadIdx.x;
    if (i >= 32 * 253 * di) return;
    int e = i % di; int l = (i / di) % 253; int b = i / (di * 253);
    const int ld = 2 * di;
    float s = cb[e];
    #pragma unroll
    for (int j = 0; j < 4; ++j) {
        int lp = l - 3 + j;
        if (lp >= 0) s += xz[((size_t)(b * 253 + lp)) * ld + e] * cw[e * 4 + j];
    }
    float r = siluf(s);
    of[i] = r; ob[i] = f2bf(r);
}

// Selective-scan: per (b,e) recurrence over L=253, h[16] in registers,
// per-step B_t/C_t vectors broadcast through LDS (shared by whole block).
__launch_bounds__(256)
__global__ void k_scan(const float* __restrict__ dt, const float* __restrict__ xm,
                       const float* __restrict__ dbl, const float* __restrict__ An,
                       const float* __restrict__ Dv, float* __restrict__ y,
                       int di, int ldd, int dtr) {
    const int b = blockIdx.y;
    const int e = blockIdx.x * 256 + threadIdx.x;
    __shared__ float sB[16], sC[16];
    float h[16];
    #pragma unroll
    for (int s = 0; s < 16; ++s) h[s] = 0.0f;
    float Ae[16];
    #pragma unroll
    for (int s = 0; s < 16; ++s) Ae[s] = An[e * 16 + s];
    float De = Dv[e];
    for (int l = 0; l < 253; ++l) {
        size_t row = (size_t)(b * 253 + l);
        if (threadIdx.x < 32) {
            int s = threadIdx.x & 15;
            float vv = dbl[row * ldd + dtr + (threadIdx.x < 16 ? s : 16 + s)];
            if (threadIdx.x < 16) sB[s] = vv; else sC[s] = vv;
        }
        __syncthreads();
        float dtv = dt[row * di + e];
        float xv  = xm[row * di + e];
        float dx  = dtv * xv;
        float accv = 0.0f;
        #pragma unroll
        for (int s = 0; s < 16; ++s) {
            float dA = __expf(dtv * Ae[s]);
            h[s] = dA * h[s] + dx * sB[s];
            accv += h[s] * sC[s];
        }
        y[row * di + e] = accv + xv * De;
        __syncthreads();
    }
}

// Gate: y2 = y * silu(z)  (z = second half of xz) -> bf16
__global__ void k_gate(const float* __restrict__ y, const float* __restrict__ xz,
                       u16* __restrict__ out, int di) {
    int i = blockIdx.x * 256 + threadIdx.x;
    if (i >= 32 * 253 * di) return;
    int e = i % di; size_t row = i / di;
    float z = xz[row * (size_t)(2 * di) + di + e];
    out[i] = f2bf(y[i] * siluf(z));
}

// ---------------------------------------------------------------------------
extern "C" void kernel_launch(void* const* d_in, const int* in_sizes, int n_in,
                              void* d_out, int out_size, void* d_ws, size_t ws_size,
                              hipStream_t stream) {
    (void)in_sizes; (void)n_in; (void)out_size; (void)ws_size;
    // input pointers (setup_inputs() dict insertion order, depth-first)
    const float* x     = (const float*)d_in[0];
    const float* enc_w = (const float*)d_in[1];
    const float* enc_b = (const float*)d_in[2];
    const float* in_w  = (const float*)d_in[3];
    const float* in_b  = (const float*)d_in[4];
    const float* out_w = (const float*)d_in[41];
    const float* out_b = (const float*)d_in[42];
    const int LBASE[2] = {5, 23};   // 18 tensors per decoder layer

    const int M   = 32 * 253;       // 8096 output rows (b,l)
    const int Mp  = 8128;           // padded to 64 (grid.y*64)
    const int NF  = 3336;
    const int NFP = 3360;           // K padded to 32

    size_t off = 0;
    auto alloc = [&](size_t bytes) -> void* {
        void* p = (char*)d_ws + off;
        off += (bytes + 255) & ~(size_t)255;
        return p;
    };

    // bf16 A-side activations: Mp rows (pad rows read, never stored)
    u16*   enc_bf = (u16*)alloc((size_t)Mp * NFP * 2);
    float* x0f    = (float*)alloc((size_t)Mp * 512 * 4);

    // packed bf16 weights (B side padded to grid.x*128 rows, K zero-padded)
    u16* inw_bf = (u16*)alloc((size_t)512 * NFP * 2);
    u16 *pwP[2], *inpj[2], *xpj[2], *dtpj[2], *opj[2];
    float* aneg[2];
    pwP[0]  = (u16*)alloc(256 * 256 * 2);   pwP[1]  = (u16*)alloc(256 * 256 * 2);
    inpj[0] = (u16*)alloc(1024 * 256 * 2);  inpj[1] = (u16*)alloc(512 * 128 * 2);
    xpj[0]  = (u16*)alloc(128 * 512 * 2);   xpj[1]  = (u16*)alloc(128 * 256 * 2);
    dtpj[0] = (u16*)alloc(512 * 32 * 2);    dtpj[1] = (u16*)alloc(256 * 32 * 2);
    opj[0]  = (u16*)alloc(256 * 512 * 2);   opj[1]  = (u16*)alloc(128 * 256 * 2);
    u16* outw_bf = (u16*)alloc(128 * 128 * 2);
    aneg[0] = (float*)alloc(512 * 16 * 4);  aneg[1] = (float*)alloc(256 * 16 * 4);

    // activation arenas, sized for layer 0, reused for layer 1
    u16*   dwT  = (u16*)alloc((size_t)32 * 256 * 256 * 2);  // (b, w, ci[256])
    float* pwf  = (float*)alloc((size_t)M * 256 * 4);
    float* xinf = (float*)alloc((size_t)M * 256 * 4);
    u16*   xinb = (u16*)alloc((size_t)Mp * 256 * 2);
    float* xzf  = (float*)alloc((size_t)M * 1024 * 4);
    float* xmf  = (float*)alloc((size_t)M * 512 * 4);
    u16*   xmb  = (u16*)alloc((size_t)Mp * 512 * 2);
    float* dblf = (float*)alloc((size_t)M * 48 * 4);
    u16*   dtA  = (u16*)alloc((size_t)Mp * 32 * 2);         // packed dt, K padded 32
    float* dtf  = (float*)alloc((size_t)M * 512 * 4);
    float* yf   = (float*)alloc((size_t)M * 512 * 4);
    u16*   y2b  = (u16*)alloc((size_t)Mp * 512 * 2);
    float* xof  = (float*)alloc((size_t)M * 256 * 4);
    u16*   xob  = (u16*)alloc((size_t)Mp * 256 * 2);

    auto g1 = [](long n) { return dim3((unsigned)((n + 255) / 256)); };
    const int GY = Mp / 64;   // 127

    // ---- weight packing (bf16, zero-padded; ~5 MB of traffic: negligible)
    k_pack<<<g1(512 * NFP), 256, 0, stream>>>(in_w, NF, 512, NF, inw_bf, NFP, 512 * NFP);
    for (int Lx = 0; Lx < 2; ++Lx) {
        int base = LBASE[Lx];
        int d = Lx ? 128 : 256, di = 2 * d, dtr = Lx ? 8 : 16;
        k_pack<<<g1(256 * 256), 256, 0, stream>>>((const float*)d_in[base + 2], 253, 253, 253,
                                                  pwP[Lx], 256, 256 * 256);
        k_pack<<<g1(2 * di * d), 256, 0, stream>>>((const float*)d_in[base + 9], d, 2 * di, d,
                                                   inpj[Lx], d, 2 * di * d);
        k_pack<<<g1(128 * di), 256, 0, stream>>>((const float*)d_in[base + 12], di, dtr + 32, di,
                                                 xpj[Lx], di, 128 * di);
        k_pack<<<g1(di * 32), 256, 0, stream>>>((const float*)d_in[base + 13], dtr, di, dtr,
                                                dtpj[Lx], 32, di * 32);
        k_pack<<<g1(d * di), 256, 0, stream>>>((const float*)d_in[base + 17], di, d, di,
                                               opj[Lx], di, d * di);
        k_aneg<<<g1(di * 16), 256, 0, stream>>>((const float*)d_in[base + 15], aneg[Lx], di * 16);
    }
    k_pack<<<g1(128 * 128), 256, 0, stream>>>(out_w, 128, 20, 128, outw_bf, 128, 128 * 128);

    // ---- encoder conv + SiLU, then 8096x512x3360 WMMA GEMM (+in_b)
    k_enc<<<g1((long)M * NFP), 256, 0, stream>>>(x, enc_w, enc_b, enc_bf);
    k_gemm<0><<<dim3(4, GY), 256, 0, stream>>>(
        enc_bf, NFP, 0, inw_bf, NFP, 0, x0f, 512, 0,
        (u16*)nullptr, in_b, (const float*)nullptr, M, 512, NFP);

    // ---- decoder layers
    const float* xin_prev = x0f;
    int Win = 512;
    for (int Lx = 0; Lx < 2; ++Lx) {
        int base = LBASE[Lx];
        int d = Lx ? 128 : 256, di = 2 * d, dtr = Lx ? 8 : 16, ldd = dtr + 32;
        int Wout = d;
        // depthwise conv (stride 2) -> transposed K-major bf16 (b, w, ci[256])
        k_dwconvT<<<g1((long)32 * Wout * 256), 256, 0, stream>>>(
            xin_prev, (const float*)d_in[base + 0], (const float*)d_in[base + 1], dwT, Win, Wout);
        // pointwise conv: batched NT GEMM  (M=253 co, N=Wout, K=256), bias per-row
        k_gemm<0><<<dim3((Wout + 127) / 128, 4, 32), 256, 0, stream>>>(
            pwP[Lx], 256, 0, dwT, 256, (size_t)Wout * 256,
            pwf, Wout, (size_t)253 * Wout,
            (u16*)nullptr, (const float*)nullptr, (const float*)d_in[base + 3], 253, Wout, 256);
        // BN + SiLU + *2 + PE
        k_bnpe<<<g1((long)M * Wout), 256, 0, stream>>>(
            pwf, (const float*)d_in[base + 4], (const float*)d_in[base + 5],
            (const float*)d_in[base + 6], (const float*)d_in[base + 7],
            (const float*)d_in[base + 8], xinf, xinb, Wout);
        // in_proj (no bias) -> xz
        k_gemm<0><<<dim3(2 * di / 128, GY), 256, 0, stream>>>(
            xinb, d, 0, inpj[Lx], d, 0, xzf, 2 * di, 0,
            (u16*)nullptr, (const float*)nullptr, (const float*)nullptr, M, 2 * di, d);
        // causal conv + SiLU -> xm (f32 + bf16)
        k_cconv<<<g1((long)M * di), 256, 0, stream>>>(
            xzf, (const float*)d_in[base + 10], (const float*)d_in[base + 11], xmf, xmb, di);
        // x_proj -> dbl (f32, ldc = dtr+32)
        k_gemm<0><<<dim3(1, GY), 256, 0, stream>>>(
            xmb, di, 0, xpj[Lx], di, 0, dblf, ldd, 0,
            (u16*)nullptr, (const float*)nullptr, (const float*)nullptr, M, ldd, di);
        // pack dt rows -> (Mp x 32) zero-padded bf16
        k_pack<<<g1((long)Mp * 32), 256, 0, stream>>>(dblf, ldd, M, dtr, dtA, 32, Mp * 32);
        // dt_proj + bias + softplus -> dt
        k_gemm<2><<<dim3(di / 128, GY), 256, 0, stream>>>(
            dtA, 32, 0, dtpj[Lx], 32, 0, dtf, di, 0,
            (u16*)nullptr, (const float*)d_in[base + 14], (const float*)nullptr, M, di, 32);
        // selective scan
        k_scan<<<dim3(di / 256, 32), 256, 0, stream>>>(
            dtf, xmf, dblf, aneg[Lx], (const float*)d_in[base + 16], yf, di, ldd, dtr);
        // gate: y * silu(z)
        k_gate<<<g1((long)M * di), 256, 0, stream>>>(yf, xzf, y2b, di);
        // out_proj -> layer output (f32 for next dw-conv, bf16 mirror for head)
        k_gemm<0><<<dim3((d + 127) / 128, GY), 256, 0, stream>>>(
            y2b, di, 0, opj[Lx], di, 0, xof, d, 0,
            (Lx == 1) ? xob : (u16*)nullptr, (const float*)nullptr, (const float*)nullptr,
            M, d, di);
        xin_prev = xof; Win = d;
    }

    // ---- head: sigmoid(x @ out_w^T + out_b) -> d_out (32,253,20) f32
    k_gemm<1><<<dim3(1, GY), 256, 0, stream>>>(
        xob, 128, 0, outw_bf, 128, 0, (float*)d_out, 20, 0,
        (u16*)nullptr, out_b, (const float*)nullptr, M, 20, 128);
}
